// Attention_19928648254023
// MI455X (gfx1250) — matrix operand order repaired and checked by reading, not compiled, not run on hardware
//
#include <hip/hip_runtime.h>
#include <hip/hip_bf16.h>

// ---------------------------------------------------------------------------
// Types for CDNA5 WMMA (wave32): v16bf A/B fragments, v8f f32 accumulator.
// ---------------------------------------------------------------------------
typedef __attribute__((ext_vector_type(16))) __bf16 v16bf;
typedef __attribute__((ext_vector_type(8)))  __bf16 v8bf;
typedef __attribute__((ext_vector_type(8)))  float  v8f;

static __device__ __forceinline__ v16bf mk16(v8bf lo, v8bf hi) {
    return __builtin_shufflevector(lo, hi, 0,1,2,3,4,5,6,7,8,9,10,11,12,13,14,15);
}
static __device__ __forceinline__ v8bf ld8(const __bf16* p) {
    return *(const v8bf*)p;   // 16-byte aligned by construction
}
static __device__ __forceinline__ v8bf cvt8(const float* p) {
    float4 a = *(const float4*)p;
    float4 b = *(const float4*)(p + 4);
    v8bf r;
    r[0] = (__bf16)a.x; r[1] = (__bf16)a.y; r[2] = (__bf16)a.z; r[3] = (__bf16)a.w;
    r[4] = (__bf16)b.x; r[5] = (__bf16)b.y; r[6] = (__bf16)b.z; r[7] = (__bf16)b.w;
    return r;
}
static __device__ __forceinline__ v8f wmma_bf16(v16bf a, v16bf b, v8f c) {
    return __builtin_amdgcn_wmma_f32_16x16x32_bf16(false, a, false, b, (short)0, c,
                                                   false, false);
}

// CDNA5 async copy: global -> LDS, 16 bytes per lane, tracked by ASYNCcnt.
static __device__ __forceinline__ void async_ld16(void* lds, const void* gptr) {
    unsigned           l = (unsigned)(uintptr_t)lds;
    unsigned long long g = (unsigned long long)(uintptr_t)gptr;
    asm volatile("global_load_async_to_lds_b128 %0, %1, off"
                 :: "v"(l), "v"(g) : "memory");
}

// ---------------------------------------------------------------------------
// Kernel 1: Wcomb = Wa @ Wk (fp32), bcomb = Wa@bk + ba; also pre-convert
// Wq/Wv/Wcomb to bf16 so the projection hot loop has no f32->bf16 work for B.
// ---------------------------------------------------------------------------
__global__ __launch_bounds__(256)
void attn_combine_kernel(const float* __restrict__ Wa, const float* __restrict__ Wk,
                         const float* __restrict__ bk, const float* __restrict__ ba,
                         const float* __restrict__ Wq, const float* __restrict__ Wv,
                         __bf16* __restrict__ Wcb, __bf16* __restrict__ Wqb,
                         __bf16* __restrict__ Wvb, float* __restrict__ bc)
{
    const int e = blockIdx.x;     // output row
    const int d = threadIdx.x;    // output col
    float acc = 0.f;
    for (int t = 0; t < 256; ++t)
        acc += Wa[e * 256 + t] * Wk[t * 256 + d];
    Wcb[e * 256 + d] = (__bf16)acc;
    Wqb[e * 256 + d] = (__bf16)Wq[e * 256 + d];
    Wvb[e * 256 + d] = (__bf16)Wv[e * 256 + d];
    if (d == 0) {
        float s = ba[e];
        for (int t = 0; t < 256; ++t) s += Wa[e * 256 + t] * bk[t];
        bc[e] = s;
    }
}

// ---------------------------------------------------------------------------
// Kernel 2: projection  out = bf16(in @ W^T + bias),  W already bf16.
//   TRANSPOSED == 0 : out row-major [16384,256].  WMMA computes out^T tiles
//     (A = W rows, B = input rows) -> 16B packed store per tile.
//   TRANSPOSED == 1 : out [B][256][2048].  WMMA computes out tiles
//     (A = input rows, B = W rows) -> 16B packed store per tile.
// ---------------------------------------------------------------------------
template <int TRANSPOSED>
__global__ __launch_bounds__(128)
void attn_proj_kernel(const float* __restrict__ in, const __bf16* __restrict__ W,
                      const float* __restrict__ bias, __bf16* __restrict__ out)
{
    const int lane = threadIdx.x & 31;
    const int wave = threadIdx.x >> 5;
    const int half = lane >> 4;
    const int l15  = lane & 15;
    const int rowBase = blockIdx.x * 64 + wave * 16;   // 16 input rows per wave

    // Fragments built once from the fp32 input rows (8 K-steps over K=256).
    v16bf inf[8];
    {
        const float* rp = in + (size_t)(rowBase + l15) * 256;
        #pragma unroll
        for (int s = 0; s < 8; ++s) {
            if (TRANSPOSED)
                inf[s] = mk16(cvt8(rp + s * 32 + half * 8),
                              cvt8(rp + s * 32 + 16 + half * 8));
            else
                inf[s] = mk16(cvt8(rp + s * 32 + half * 16),
                              cvt8(rp + s * 32 + half * 16 + 8));
        }
    }

    #pragma unroll 1
    for (int et = 0; et < 16; ++et) {
        const __bf16* wp = W + (size_t)(et * 16 + l15) * 256;  // W row for lane
        v8f acc = {};
        #pragma unroll
        for (int s = 0; s < 8; ++s) {
            if (TRANSPOSED) {
                v16bf bf = mk16(ld8(wp + s * 32 + half * 8),
                                ld8(wp + s * 32 + 16 + half * 8));
                acc = wmma_bf16(inf[s], bf, acc);
            } else {
                v16bf af = mk16(ld8(wp + s * 32 + half * 16),
                                ld8(wp + s * 32 + half * 16 + 8));
                acc = wmma_bf16(af, inf[s], acc);
            }
        }
        if (TRANSPOSED) {
            const int e  = et * 16 + l15;
            const float bv = bias[e];
            v8bf pk;
            #pragma unroll
            for (int r = 0; r < 8; ++r) pk[r] = (__bf16)(acc[r] + bv);
            const int b  = rowBase >> 11;                // Sk = 2048, 64 | 2048
            const int sk = (rowBase & 2047) + 8 * half;  // 8 consecutive sk
            *(v8bf*)&out[((size_t)(b * 256 + e)) * 2048 + sk] = pk;
        } else {
            const float4 b0 = *(const float4*)(bias + et * 16 + 8 * half);
            const float4 b1 = *(const float4*)(bias + et * 16 + 8 * half + 4);
            v8bf pk;
            pk[0] = (__bf16)(acc[0] + b0.x); pk[1] = (__bf16)(acc[1] + b0.y);
            pk[2] = (__bf16)(acc[2] + b0.z); pk[3] = (__bf16)(acc[3] + b0.w);
            pk[4] = (__bf16)(acc[4] + b1.x); pk[5] = (__bf16)(acc[5] + b1.y);
            pk[6] = (__bf16)(acc[6] + b1.z); pk[7] = (__bf16)(acc[7] + b1.w);
            *(v8bf*)&out[(size_t)(rowBase + l15) * 256 + et * 16 + 8 * half] = pk;
        }
    }
}

// ---------------------------------------------------------------------------
// Kernel 3: flash attention with double-buffered async global->LDS staging.
//   qh, wkh : [B*2048, 256] bf16.   vhT : [B][256][2048] bf16.
//   out     : [B*2048, 256] fp32.
// Block = 128 threads (4 waves) = 64 q-rows; K chunks of 32, ping-pong LDS.
// q A-fragments are re-materialized from the L0-hot q rows each chunk with
// single-step liveness so the wave stays under 256 VGPRs (no MSB banking).
// ---------------------------------------------------------------------------
#define KT_STRIDE 264   // 256 + 8 pad (bf16 elems), 16B-aligned rows
#define VT_STRIDE 40    // 32 + 8 pad

__global__ __launch_bounds__(128)
void attn_flash_kernel(const __bf16* __restrict__ qh, const __bf16* __restrict__ wkh,
                       const __bf16* __restrict__ vhT, float* __restrict__ out)
{
    __shared__ __align__(16) __bf16 KtD[2][32 * KT_STRIDE];   // 2 x 16896 B
    __shared__ __align__(16) __bf16 VtD[2][256 * VT_STRIDE];  // 2 x 20480 B
    __shared__ __align__(16) __bf16 Pst[4][16 * VT_STRIDE];   // 5120 B

    const int tid  = threadIdx.x;
    const int lane = tid & 31;
    const int wave = tid >> 5;
    const int half = lane >> 4;
    const int l15  = lane & 15;

    const int b    = blockIdx.x >> 5;           // 32 q-blocks per batch
    const int qblk = (blockIdx.x & 31) * 64;
    const size_t qrow0 = (size_t)b * 2048 + qblk + wave * 16;

    const __bf16* kbase = wkh + (size_t)b * 2048 * 256;
    const __bf16* vbase = vhT + (size_t)b * 256 * 2048;
    const __bf16* qp    = qh + (qrow0 + l15) * 256;   // this lane's q row

    // issue one chunk's tiles (16 async b128 per thread: 8 for K, 8 for V^T)
    auto issue_tiles = [&](int kc, __bf16* KtB, __bf16* VtB) {
        #pragma unroll
        for (int it = 0; it < 8; ++it) {
            const int i = tid + it * 128;                 // 0..1023 x 16B
            const int r = i >> 5, c = (i & 31) * 8;
            async_ld16(&KtB[r * KT_STRIDE + c],
                       &kbase[(size_t)(kc + r) * 256 + c]);
        }
        #pragma unroll
        for (int it = 0; it < 8; ++it) {
            const int i = tid + it * 128;
            const int j = i >> 2, c = (i & 3) * 8;
            async_ld16(&VtB[j * VT_STRIDE + c],
                       &vbase[(size_t)j * 2048 + kc + c]);
        }
    };

    v8f Oacc[16];
    #pragma unroll
    for (int j = 0; j < 16; ++j) Oacc[j] = (v8f){};
    float rm[8], rs[8];
    #pragma unroll
    for (int r = 0; r < 8; ++r) { rm[r] = -3.0e38f; rs[r] = 0.f; }

    issue_tiles(0, KtD[0], VtD[0]);             // prologue: chunk 0 in flight

    int ib = 0;
    #pragma unroll 1
    for (int kc = 0; kc < 2048; kc += 32, ib ^= 1) {
        const bool hasNext = (kc + 32) < 2048;
        if (hasNext) {
            issue_tiles(kc + 32, KtD[ib ^ 1], VtD[ib ^ 1]);
            // 32 outstanding, in-order completion: <=16 left => chunk kc done
            asm volatile("s_wait_asynccnt 16" ::: "memory");
        } else {
            asm volatile("s_wait_asynccnt 0" ::: "memory");
        }
        __syncthreads();                         // publish current tiles
        const __bf16* Kt = KtD[ib];
        const __bf16* Vt = VtD[ib];

        // --- S = q @ wk^T : 16 x 32 fp32, s-major so each q A-fragment lives
        //     for exactly one step (two WMMAs) ---------------------------------
        v8f S0 = {}, S1 = {};
        {
            const __bf16* kp0 = &Kt[l15 * KT_STRIDE];
            const __bf16* kp1 = &Kt[(16 + l15) * KT_STRIDE];
            #pragma unroll
            for (int s = 0; s < 8; ++s) {
                v16bf a  = mk16(ld8(qp + s * 32 + half * 16),
                                ld8(qp + s * 32 + half * 16 + 8));
                v16bf b0 = mk16(ld8(kp0 + s * 32 + half * 16),
                                ld8(kp0 + s * 32 + half * 16 + 8));
                v16bf b1 = mk16(ld8(kp1 + s * 32 + half * 16),
                                ld8(kp1 + s * 32 + half * 16 + 8));
                S0 = wmma_bf16(a, b0, S0);
                S1 = wmma_bf16(a, b1, S1);
            }
        }

        // --- online softmax (rows live in 16-lane halves) -------------------
        float scl[8];
        #pragma unroll
        for (int r = 0; r < 8; ++r) {
            float mx = fmaxf(S0[r], S1[r]);
            #pragma unroll
            for (int m = 8; m >= 1; m >>= 1)
                mx = fmaxf(mx, __shfl_xor(mx, m, 32));
            const float nm = fmaxf(rm[r], mx);
            scl[r] = __expf(rm[r] - nm);
            rm[r]  = nm;
        }
        #pragma unroll
        for (int r = 0; r < 8; ++r) {
            const float p0 = __expf(S0[r] - rm[r]);
            const float p1 = __expf(S1[r] - rm[r]);
            S0[r] = p0; S1[r] = p1;
            float sm = p0 + p1;
            #pragma unroll
            for (int m = 8; m >= 1; m >>= 1)
                sm += __shfl_xor(sm, m, 32);
            rs[r] = rs[r] * scl[r] + sm;
        }
        #pragma unroll
        for (int j = 0; j < 16; ++j)
            #pragma unroll
            for (int r = 0; r < 8; ++r)
                Oacc[j][r] *= scl[r];

        // --- P (C layout) -> per-wave LDS -> A layout -----------------------
        // Same-wave LDS ops complete in order (ISA 7.3): compiler fence only.
        __bf16* pw = &Pst[wave][0];
        #pragma unroll
        for (int r = 0; r < 8; ++r) {
            pw[(r + 8 * half) * VT_STRIDE + l15]      = (__bf16)S0[r];
            pw[(r + 8 * half) * VT_STRIDE + 16 + l15] = (__bf16)S1[r];
        }
        asm volatile("" ::: "memory");           // keep ds_store before ds_load

        // --- O += P @ V : 16 output tiles, one 16x16x32 WMMA each -----------
        const __bf16* pr = &Pst[wave][l15 * VT_STRIDE];
        const v16bf pf = mk16(ld8(pr + half * 16), ld8(pr + half * 16 + 8));
        #pragma unroll
        for (int j = 0; j < 16; ++j) {
            const __bf16* vp = &Vt[(j * 16 + l15) * VT_STRIDE];
            v16bf bf = mk16(ld8(vp + half * 16), ld8(vp + half * 16 + 8));
            Oacc[j] = wmma_bf16(pf, bf, Oacc[j]);
        }
        __syncthreads();   // all reads of current tiles done before refill
    }

    // --- epilogue: normalize and store fp32 ---------------------------------
    #pragma unroll
    for (int r = 0; r < 8; ++r) rs[r] = 1.0f / rs[r];
    #pragma unroll
    for (int j = 0; j < 16; ++j) {
        #pragma unroll
        for (int r = 0; r < 8; ++r) {
            const int m = r + 8 * half;
            out[(qrow0 + m) * 256 + j * 16 + l15] = Oacc[j][r] * rs[r];
        }
    }
}

// ---------------------------------------------------------------------------
// Launch: combine+convert weights, 3 WMMA projections, flash attention.
// ---------------------------------------------------------------------------
extern "C" void kernel_launch(void* const* d_in, const int* in_sizes, int n_in,
                              void* d_out, int out_size, void* d_ws, size_t ws_size,
                              hipStream_t stream) {
    const float* x      = (const float*)d_in[0];
    const float* states = (const float*)d_in[1];
    const float* Wq = (const float*)d_in[2];
    const float* bq = (const float*)d_in[3];
    const float* Wk = (const float*)d_in[4];
    const float* bk = (const float*)d_in[5];
    const float* Wv = (const float*)d_in[6];
    const float* bv = (const float*)d_in[7];
    const float* Wa = (const float*)d_in[8];
    const float* ba = (const float*)d_in[9];
    float* outp = (float*)d_out;

    // Workspace layout (16B aligned):
    //   Wqb/Wcb/Wvb bf16 [256*256]    3 x 131072 B
    //   bc  fp32 [256]                      1024 B
    //   qh  bf16 [16384*256]             8388608 B
    //   wkh bf16 [16384*256]             8388608 B
    //   vhT bf16 [8][256][2048]          8388608 B   (~25.6 MB total)
    char*   ws  = (char*)d_ws;
    __bf16* Wqb = (__bf16*)ws;
    __bf16* Wcb = (__bf16*)(ws + 131072);
    __bf16* Wvb = (__bf16*)(ws + 262144);
    float*  bc  = (float*)(ws + 393216);
    __bf16* qh  = (__bf16*)(ws + 394240);
    __bf16* wkh = qh  + (size_t)16384 * 256;
    __bf16* vhT = wkh + (size_t)16384 * 256;

    attn_combine_kernel<<<256, 256, 0, stream>>>(Wa, Wk, bk, ba, Wq, Wv,
                                                 Wcb, Wqb, Wvb, bc);
    attn_proj_kernel<0><<<256, 128, 0, stream>>>(x,      Wqb, bq, qh);
    attn_proj_kernel<0><<<256, 128, 0, stream>>>(states, Wcb, bc, wkh);
    attn_proj_kernel<1><<<256, 128, 0, stream>>>(states, Wvb, bv, vhT);
    attn_flash_kernel<<<256, 128, 0, stream>>>(qh, wkh, vhT, outp);
}
